// GatherModel_39582418600429
// MI455X (gfx1250) — matrix operand order repaired and checked by reading, not compiled, not run on hardware
//
#include <hip/hip_runtime.h>

#define N_NODES 50000
#define N_EDGES 150000
#define D       42
#define E_IN    10
#define N_STEPS 6

#define KPAD 64            // K=42 padded to 2 x 32 for wmma_f32_16x16x32_f16
#define NT   112           // 1764 -> 112 tiles of 16
#define NPAD (NT * 16)     // 1792, row stride of w_e
#define MT   (N_EDGES / 16) // 9375 M-tiles, exact

typedef _Float16 v8h  __attribute__((ext_vector_type(8)));
typedef _Float16 v16h __attribute__((ext_vector_type(16)));
typedef float    v8f  __attribute__((ext_vector_type(8)));

// ---------------- one-time kernels ----------------

// out0 = relu(n_feat @ lin0_w + lin0_b)   [N,42]@[42,42] -- trivial VALU
__global__ void k_lin0(const float* __restrict__ nf, const float* __restrict__ w,
                       const float* __restrict__ b, float* __restrict__ out) {
    int gid = blockIdx.x * blockDim.x + threadIdx.x;
    if (gid >= N_NODES * D) return;
    int n = gid / D, o = gid - n * D;
    const float* row = nf + n * D;
    float acc = b[o];
#pragma unroll
    for (int d = 0; d < D; ++d) acc = fmaf(row[d], w[d * D + o], acc);
    out[gid] = fmaxf(acc, 0.f);
}

// h_e = relu(e_feat @ en1_w + en1_b), written padded to 64 f16 per edge (A operand)
__global__ void k_edge_h(const float* __restrict__ ef, const float* __restrict__ w1,
                         const float* __restrict__ b1, _Float16* __restrict__ hpad) {
    int gid = blockIdx.x * blockDim.x + threadIdx.x;
    if (gid >= N_EDGES * KPAD) return;
    int e = gid >> 6, j = gid & 63;
    float v = 0.f;
    if (j < D) {
        v = b1[j];
        const float* row = ef + e * E_IN;
#pragma unroll
        for (int i = 0; i < E_IN; ++i) v = fmaf(row[i], w1[i * D + j], v);
        v = fmaxf(v, 0.f);
    }
    hpad[gid] = (_Float16)v;
}

// Pre-swizzle en2_w into this kernel's B-operand layout (contiguous k, same order as A):
// Bp[((nt*2+kb)*32 + lane)*16 + j] = B[k = kb*32 + (lane>>4)*16 + j][n = nt*16 + (lane&15)]
// so the GEMM kernel loads each lane's v16h as one contiguous 32-byte chunk.
__global__ void k_pack_B(const float* __restrict__ w2, const float* __restrict__ b2,
                         _Float16* __restrict__ Bp, float* __restrict__ b2pad) {
    const int total = NT * 2 * 32 * 16;
    for (int idx = blockIdx.x * blockDim.x + threadIdx.x; idx < total;
         idx += gridDim.x * blockDim.x) {
        int j    = idx & 15;
        int lane = (idx >> 4) & 31;
        int kb   = (idx >> 9) & 1;
        int nt   = idx >> 10;
        int k = kb * 32 + ((lane >> 4) << 4) + j;
        int n = nt * 16 + (lane & 15);
        float v = (k < D && n < D * D) ? w2[k * (D * D) + n] : 0.f;
        Bp[idx] = (_Float16)v;
    }
    for (int n = blockIdx.x * blockDim.x + threadIdx.x; n < NPAD;
         n += gridDim.x * blockDim.x)
        b2pad[n] = (n < D * D) ? b2[n] : 0.f;
}

// w_e = h_e @ en2_w + en2_b  -> f16, row stride NPAD.
// One wave per 16-edge M-tile; A lives in 16 VGPRs across all 112 N-tiles.
// 2 x v_wmma_f32_16x16x32_f16 per 16x16 tile (K=64 padded).
__global__ void __launch_bounds__(256)
k_edge_weights(const _Float16* __restrict__ hpad, const _Float16* __restrict__ Bp,
               const float* __restrict__ b2pad, _Float16* __restrict__ W) {
    int wave = threadIdx.x >> 5;
    int lane = threadIdx.x & 31;
    int mt = blockIdx.x * 8 + wave;
    if (mt >= MT) return;                 // wave-uniform: EXEC all-1s inside
    int r = lane & 15, half = lane >> 4;

    // A layout (matches Bp): lane half h -> elems 0..15 = K 16h..16h+15.
    // One contiguous 32-byte chunk per kb.
    const _Float16* arow = hpad + (size_t)(mt * 16 + r) * KPAD;
    v16h a0 = *(const v16h*)(arow + half * 16);
    v16h a1 = *(const v16h*)(arow + 32 + half * 16);

    size_t rowbase = (size_t)mt * 16 * NPAD;
    for (int nt = 0; nt < NT; ++nt) {
        v16h b0 = *(const v16h*)(Bp + ((size_t)(nt * 2 + 0) * 32 + lane) * 16);
        v16h b1 = *(const v16h*)(Bp + ((size_t)(nt * 2 + 1) * 32 + lane) * 16);
        v8f c = {};
        c = __builtin_amdgcn_wmma_f32_16x16x32_f16(false, a0, false, b0,
                                                   (short)0, c, false, false);
        c = __builtin_amdgcn_wmma_f32_16x16x32_f16(false, a1, false, b1,
                                                   (short)0, c, false, false);
        int nbase = nt * 16;
        float bias = b2pad[nbase + r];
        // C layout: VGPR v -> row v + 8*half, col r
        _Float16* outp = W + rowbase + (size_t)half * 8 * NPAD + nbase + r;
#pragma unroll
        for (int v = 0; v < 8; ++v)
            outp[(size_t)v * NPAD] = (_Float16)(c[v] + bias);
    }
}

// ---------------- per-step kernels ----------------

__global__ void k_zero(float* __restrict__ p, int n) {
    int g = blockIdx.x * blockDim.x + threadIdx.x;
    if (g < n) p[g] = 0.f;
}

// msg[e,o] = sum_d out[src[e],d] * w_e[e,d,o];  atomically scattered to agg[dst[e],o]
// Streaming-read bound (~0.54 GB f16 / step).
__global__ void k_msg(const float* __restrict__ out, const _Float16* __restrict__ W,
                      const int* __restrict__ src, const int* __restrict__ dst,
                      float* __restrict__ agg) {
    int gid = blockIdx.x * blockDim.x + threadIdx.x;
    if (gid >= N_EDGES * D) return;
    int e = gid / D, o = gid - e * D;
    const float* x = out + (size_t)src[e] * D;
    const _Float16* w = W + (size_t)e * NPAD + o;   // logical index d*42+o
    float acc = 0.f;
#pragma unroll 6
    for (int d = 0; d < D; ++d) acc = fmaf(x[d], (float)w[d * D], acc);
    atomicAdd(agg + (size_t)dst[e] * D + o, acc);
}

// m = relu(agg + out + conv_bias); out_new = [m,out] @ msg_w + msg_b
__global__ void k_update(const float* __restrict__ agg, const float* __restrict__ outc,
                         const float* __restrict__ conv_bias,
                         const float* __restrict__ msg_w, const float* __restrict__ msg_b,
                         float* __restrict__ outn) {
    int gid = blockIdx.x * blockDim.x + threadIdx.x;
    if (gid >= N_NODES * D) return;
    int n = gid / D, o = gid - n * D;
    const float* ar = agg + (size_t)n * D;
    const float* orow = outc + (size_t)n * D;
    float acc = msg_b[o];
#pragma unroll
    for (int d = 0; d < D; ++d) {
        float m = fmaxf(ar[d] + orow[d] + conv_bias[d], 0.f);
        acc = fmaf(m, msg_w[d * D + o], acc);
        acc = fmaf(orow[d], msg_w[(D + d) * D + o], acc);
    }
    outn[gid] = acc;
}

__global__ void k_final(const float* __restrict__ outc, const float* __restrict__ nf,
                        float* __restrict__ dout) {
    int g = blockIdx.x * blockDim.x + threadIdx.x;
    if (g < N_NODES * D) dout[g] = outc[g] + nf[g];
}

// ---------------- host launcher ----------------

extern "C" void kernel_launch(void* const* d_in, const int* in_sizes, int n_in,
                              void* d_out, int out_size, void* d_ws, size_t ws_size,
                              hipStream_t stream) {
    const float* n_feat    = (const float*)d_in[0];
    const float* e_feat    = (const float*)d_in[1];
    const int*   src       = (const int*)d_in[2];
    const int*   dst       = (const int*)d_in[3];
    const float* lin0_w    = (const float*)d_in[4];
    const float* lin0_b    = (const float*)d_in[5];
    const float* msg_w     = (const float*)d_in[6];
    const float* msg_b     = (const float*)d_in[7];
    const float* en1_w     = (const float*)d_in[8];
    const float* en1_b     = (const float*)d_in[9];
    const float* en2_w     = (const float*)d_in[10];
    const float* en2_b     = (const float*)d_in[11];
    const float* conv_bias = (const float*)d_in[12];

    char* ws = (char*)d_ws;
    size_t off = 0;
    auto take = [&](size_t bytes) -> char* {
        char* p = ws + off;
        off = (off + bytes + 255) & ~(size_t)255;
        return p;
    };
    _Float16* hpad  = (_Float16*)take((size_t)N_EDGES * KPAD * 2);
    _Float16* Bp    = (_Float16*)take((size_t)NT * 2 * 32 * 16 * 2);
    float*    b2pad = (float*)take((size_t)NPAD * 4);
    _Float16* Wf    = (_Float16*)take((size_t)N_EDGES * NPAD * 2);   // 537.6 MB
    float*    outA  = (float*)take((size_t)N_NODES * D * 4);
    float*    outB  = (float*)take((size_t)N_NODES * D * 4);
    float*    agg   = (float*)take((size_t)N_NODES * D * 4);

    const int TB = 256;
    const int nd_grid = (N_NODES * D + TB - 1) / TB;
    const int ed_grid = (N_EDGES * D + TB - 1) / TB;

    // one-time: node embed, edge MLP layer 1 (padded f16), B pre-swizzle, big WMMA GEMM
    k_lin0<<<nd_grid, TB, 0, stream>>>(n_feat, lin0_w, lin0_b, outA);
    k_edge_h<<<(N_EDGES * KPAD + TB - 1) / TB, TB, 0, stream>>>(e_feat, en1_w, en1_b, hpad);
    k_pack_B<<<64, TB, 0, stream>>>(en2_w, en2_b, Bp, b2pad);
    k_edge_weights<<<(MT + 7) / 8, TB, 0, stream>>>(hpad, Bp, b2pad, Wf);

    // 6 message-passing steps, ping-pong node state
    float* cur = outA;
    float* nxt = outB;
    for (int s = 0; s < N_STEPS; ++s) {
        k_zero<<<nd_grid, TB, 0, stream>>>(agg, N_NODES * D);
        k_msg<<<ed_grid, TB, 0, stream>>>(cur, Wf, src, dst, agg);
        k_update<<<nd_grid, TB, 0, stream>>>(agg, cur, conv_bias, msg_w, msg_b, nxt);
        float* t = cur; cur = nxt; nxt = t;
    }
    k_final<<<nd_grid, TB, 0, stream>>>(cur, n_feat, (float*)d_out);
}
